// DeformBottleneck_71605694759689
// MI455X (gfx1250) — matrix operand order repaired and checked by reading, not compiled, not run on hardware
//
#include <hip/hip_runtime.h>
#include <hip/hip_bf16.h>

typedef __attribute__((ext_vector_type(16))) __bf16 v16bf;
typedef __attribute__((ext_vector_type(8)))  float  v8f;

union FragBF {
    v16bf v;
    uint4 q[2];
    unsigned short h[16];
};

__device__ inline unsigned short f2bf(float f) {
    unsigned int u = __float_as_uint(f);
    unsigned int r = u + 0x7FFFu + ((u >> 16) & 1u);
    return (unsigned short)(r >> 16);
}
__device__ inline float bf2f(unsigned short h) {
    return __uint_as_float(((unsigned int)h) << 16);
}

// A-fragment (16x32 bf16, row-major [M,K] source):
// lane: m = lane&15, half = lane>>4
// elems e = A[m][kb + 16*half + e]  (16 contiguous bf16)
__device__ inline v16bf load_a_global(const unsigned short* __restrict__ W,
                                      int K, int m, int kb, int half) {
    FragBF f;
    const unsigned short* p = W + (size_t)m * K + kb + 16 * half;
    f.q[0] = *(const uint4*)(p);
    f.q[1] = *(const uint4*)(p + 8);
    return f.v;
}

// B-fragment (32x16 bf16) from LDS row laid out [n][k]:
// lane: n = lane&15 (row base passed in), half = lane>>4
// elems e = B[kb + 16*half + e]  (16 contiguous bf16 = 32B)
__device__ inline v16bf load_b_lds(const unsigned short* Brow, int kb, int half) {
    FragBF f;
    const unsigned short* p = Brow + kb + 16 * half;
    f.q[0] = *(const uint4*)(p);
    f.q[1] = *(const uint4*)(p + 8);
    return f.v;
}

#define WMMA_BF16(a, b, c) \
    __builtin_amdgcn_wmma_f32_16x16x32_bf16(false, (a), false, (b), (short)0, (c), false, false)

// Per-lane prefetch of a wave's 16 A-rows (2 cachelines each per call).
// lanes 0-15 -> rows, lanes 16-31 -> rows at +128B. Lowers to global_prefetch_b8.
__device__ inline void prefetch_a(const unsigned short* W, int K, int m0,
                                  int nl, int half, int byte_off) {
    const char* p = (const char*)(W + (size_t)(m0 + nl) * K) + half * 128 + byte_off;
    __builtin_prefetch(p, 0, 0);
}

// ---------------- prep: fp32 -> bf16 weight conversion ----------------
__global__ void cvt_f32_bf16(const float* __restrict__ src,
                             unsigned short* __restrict__ dst, int n) {
    int i = blockIdx.x * 256 + threadIdx.x;
    if (i < n) dst[i] = f2bf(src[i]);
}
__global__ void fill_zero_u16(unsigned short* __restrict__ dst, int n) {
    int i = blockIdx.x * 256 + threadIdx.x;
    if (i < n) dst[i] = 0;
}

// ---------------- k1: fused conv1(+bn+relu x2) and downsample(+bn), N=64 ----------------
// x:[4,256,16384] f32 -> t:[4,128,16384] bf16, ident:[4,128,16384] f32
__global__ __launch_bounds__(256) void k1(
    const float* __restrict__ x,
    const unsigned short* __restrict__ w1b,
    const unsigned short* __restrict__ wdsb,
    const float* __restrict__ s1a, const float* __restrict__ t1a,
    const float* __restrict__ s1b, const float* __restrict__ t1b,
    const float* __restrict__ bds, const float* __restrict__ sd,
    const float* __restrict__ td,
    unsigned short* __restrict__ tout, float* __restrict__ ident) {
    __shared__ unsigned short Bt[64 * 272];   // [n][k], K=256 padded to 272 (34,816 B)
    int gp  = blockIdx.x * 64;
    int b   = gp >> 14;
    int pix = gp & 16383;
    const float* xb = x + (size_t)b * 4194304 + pix;
    int tid = threadIdx.x;
    int wave = tid >> 5, lane = tid & 31;
    int half = lane >> 4, nl = lane & 15;
    int m0 = wave * 16;
    // warm the A streams (w1 row = 512B -> 2 calls each)
    prefetch_a(w1b,  256, m0, nl, half, 0);
    prefetch_a(w1b,  256, m0, nl, half, 256);
    prefetch_a(wdsb, 256, m0, nl, half, 0);
    prefetch_a(wdsb, 256, m0, nl, half, 256);
    for (int j = 0; j < 64; ++j) {
        int idx = tid + j * 256;              // 64*256 elements
        int n = idx & 63, c = idx >> 6;
        Bt[n * 272 + c] = f2bf(xb[(size_t)c * 16384 + n]);
    }
    __syncthreads();
    v8f acc1[4] = {}, acc2[4] = {};
    for (int s = 0; s < 8; ++s) {
        int kb = s * 32;
        v16bf a1 = load_a_global(w1b,  256, m0 + nl, kb, half);
        v16bf a2 = load_a_global(wdsb, 256, m0 + nl, kb, half);
#pragma unroll
        for (int g = 0; g < 4; ++g) {
            v16bf bfr = load_b_lds(Bt + (g * 16 + nl) * 272, kb, half);
            acc1[g] = WMMA_BF16(a1, bfr, acc1[g]);
            acc2[g] = WMMA_BF16(a2, bfr, acc2[g]);
        }
    }
    size_t ob0 = (size_t)b * 2097152 + pix + nl;
#pragma unroll
    for (int r = 0; r < 8; ++r) {
        int m = m0 + r + 8 * half;
        float sa = s1a[m], ta = t1a[m], sb = s1b[m], tb2 = t1b[m];
        float bd = bds[m], sdd = sd[m], tdd = td[m];
        size_t mb = ob0 + (size_t)m * 16384;
#pragma unroll
        for (int g = 0; g < 4; ++g) {
            float v = acc1[g][r];
            v = fmaxf(sa * v + ta, 0.f);
            v = fmaxf(sb * v + tb2, 0.f);
            tout[mb + g * 16] = f2bf(v);
            ident[mb + g * 16] = sdd * (acc2[g][r] + bd) + tdd;
        }
    }
}

// ---------------- k2: 3x3 offset conv (M=27 padded to 32, K=1152, N=64) ----------------
// t bf16 -> off:[4,27,16384] f32 (with bias)
__global__ __launch_bounds__(256) void k2(
    const unsigned short* __restrict__ t,
    const unsigned short* __restrict__ woffb,
    const float* __restrict__ boff,
    float* __restrict__ off) {
    __shared__ unsigned short Halo[128 * 3 * 72];  // [i][ky][xi], xi: 66 used, pad 72
    int gp  = blockIdx.x * 64;
    int b   = gp >> 14;
    int pix = gp & 16383;
    int y = pix >> 7, x0 = pix & 127;              // tile = 64 contiguous x in one row
    const unsigned short* tb = t + (size_t)b * 2097152;
    int tid = threadIdx.x;
    int wave = tid >> 5, lane = tid & 31;
    int half = lane >> 4, nl = lane & 15;
    int mt = wave & 1, ns = wave >> 1;             // 2 M-tiles x 4 N-subtiles
    int m0 = mt * 16, n0 = ns * 16;
    prefetch_a(woffb, 1152, m0, nl, half, 0);
    prefetch_a(woffb, 1152, m0, nl, half, 256);
    for (int j = 0; j < 99; ++j) {                 // 128*3*66 = 25344 elements
        int idx = tid + j * 256;
        if (idx < 25344) {
            int xi = idx % 66;
            int rest = idx / 66;
            int ky = rest % 3;
            int i  = rest / 3;
            int yy = y + ky - 1;
            int xx = x0 + xi - 1;
            unsigned short v = 0;
            if ((unsigned)yy < 128u && (unsigned)xx < 128u)
                v = tb[(size_t)i * 16384 + yy * 128 + xx];
            Halo[(i * 3 + ky) * 72 + xi] = v;
        }
    }
    __syncthreads();
    int nn = n0 + nl;
    v8f acc = {};
    for (int s = 0; s < 36; ++s) {
        int kb = s * 32;
        FragBF fb;
#pragma unroll
        for (int e = 0; e < 16; ++e) {
            int k = kb + 16 * half + e;
            int i = k / 9, r = k % 9;
            fb.h[e] = Halo[(i * 3 + r / 3) * 72 + nn + (r % 3)];
        }
        v16bf a = load_a_global(woffb, 1152, m0 + nl, kb, half);
        acc = WMMA_BF16(a, fb.v, acc);
    }
#pragma unroll
    for (int r = 0; r < 8; ++r) {
        int m = m0 + r + 8 * half;
        if (m < 27)
            off[((size_t)b * 27 + m) * 16384 + pix + n0 + nl] = acc[r] + boff[m];
    }
}

// ---------------- k3: DCNv2 deform conv (M=128, K=1152, N=32) + bn2 + relu ----------------
// t bf16, off f32 -> u:[4,128,16384] bf16
__global__ __launch_bounds__(256) void k3(
    const unsigned short* __restrict__ t,
    const float* __restrict__ off,
    const unsigned short* __restrict__ wdcb,
    const float* __restrict__ bdc,
    const float* __restrict__ s2, const float* __restrict__ t2,
    unsigned short* __restrict__ u) {
    __shared__ unsigned short Bt[32 * 1160];       // [n][k], K=1152 pad 1160 (74,240 B)
    __shared__ short cy0[288], cx0[288];
    __shared__ float cw0[288], cw1[288], cw2[288], cw3[288];
    int gp  = blockIdx.x * 32;
    int b   = gp >> 14;
    int pix = gp & 16383;
    int y = pix >> 7, x0 = pix & 127;              // 32 contiguous x in one row
    const unsigned short* tb = t + (size_t)b * 2097152;
    const float* offb = off + (size_t)b * 27 * 16384 + pix;
    int tid = threadIdx.x;
    int wave = tid >> 5, lane = tid & 31;
    int half = lane >> 4, nl = lane & 15;
    int m0 = wave * 16;
    // warm first K-steps of this wave's w_dc rows
    prefetch_a(wdcb, 1152, m0, nl, half, 0);
    prefetch_a(wdcb, 1152, m0, nl, half, 256);
    prefetch_a(wdcb, 1152, m0, nl, half, 512);
    for (int e = tid; e < 288; e += 256) {         // (pixel n, tap kk) offset decode
        int n = e & 31, kk = e >> 5;
        int kyr = kk / 3, kxr = kk % 3;
        float dy = offb[(size_t)kk * 16384 + n];
        float dx = offb[(size_t)(9 + kk) * 16384 + n];
        float ml = offb[(size_t)(18 + kk) * 16384 + n];
        float mask = 1.f / (1.f + __expf(-ml));
        float py = (float)(y + kyr - 1) + dy;
        float px = (float)(x0 + n + kxr - 1) + dx;
        float fy = floorf(py), fx = floorf(px);
        float wy = py - fy, wx = px - fx;
        cy0[e] = (short)(int)fy;
        cx0[e] = (short)(int)fx;
        cw0[e] = (1.f - wy) * (1.f - wx) * mask;   // mask folded into corner weights
        cw1[e] = (1.f - wy) * wx * mask;
        cw2[e] = wy * (1.f - wx) * mask;
        cw3[e] = wy * wx * mask;
    }
    __syncthreads();
    for (int j = 0; j < 144; ++j) {                // build 1152x32 sampled B tile
        int idx = tid + j * 256;                   // 36864 elements
        int n = idx & 31;
        int rest = idx >> 5;                       // < 1152
        int kk = rest % 9;
        int i  = rest / 9;
        int p = kk * 32 + n;
        int y0 = cy0[p], xx = cx0[p];
        int base = i * 16384;
        float acc = 0.f;
        if ((unsigned)y0 < 128u) {
            int rb = base + y0 * 128;
            if ((unsigned)xx < 128u)       acc += bf2f(tb[rb + xx])     * cw0[p];
            if ((unsigned)(xx + 1) < 128u) acc += bf2f(tb[rb + xx + 1]) * cw1[p];
        }
        if ((unsigned)(y0 + 1) < 128u) {
            int rb = base + (y0 + 1) * 128;
            if ((unsigned)xx < 128u)       acc += bf2f(tb[rb + xx])     * cw2[p];
            if ((unsigned)(xx + 1) < 128u) acc += bf2f(tb[rb + xx + 1]) * cw3[p];
        }
        Bt[n * 1160 + i * 9 + kk] = f2bf(acc);
    }
    __syncthreads();
    v8f acc[2] = {};
    for (int s = 0; s < 36; ++s) {
        int kb = s * 32;
        v16bf a = load_a_global(wdcb, 1152, m0 + nl, kb, half);
#pragma unroll
        for (int g = 0; g < 2; ++g) {
            v16bf bfr = load_b_lds(Bt + (g * 16 + nl) * 1160, kb, half);
            acc[g] = WMMA_BF16(a, bfr, acc[g]);
        }
    }
    size_t ob0 = (size_t)b * 2097152 + pix + nl;
#pragma unroll
    for (int r = 0; r < 8; ++r) {
        int m = m0 + r + 8 * half;
        float bd = bdc[m], s2m = s2[m], t2m = t2[m];
        size_t mb = ob0 + (size_t)m * 16384;
#pragma unroll
        for (int g = 0; g < 2; ++g) {
            float v = acc[g][r] + bd;
            v = fmaxf(s2m * v + t2m, 0.f);
            u[mb + g * 16] = f2bf(v);
        }
    }
}

// ---------------- k4: conv3 (M=128,K=128,N=64) + bn3a+relu + bn3b + ident + relu ----------------
__global__ __launch_bounds__(256) void k4(
    const unsigned short* __restrict__ u,
    const unsigned short* __restrict__ w3b,
    const float* __restrict__ s3a, const float* __restrict__ t3a,
    const float* __restrict__ s3b, const float* __restrict__ t3b,
    const float* __restrict__ ident,
    float* __restrict__ out) {
    __shared__ unsigned short Bt[64 * 136];        // [n][k], K=128 padded (17,408 B)
    int gp  = blockIdx.x * 64;
    int b   = gp >> 14;
    int pix = gp & 16383;
    const unsigned short* ub = u + (size_t)b * 2097152 + pix;
    int tid = threadIdx.x;
    int wave = tid >> 5, lane = tid & 31;
    int half = lane >> 4, nl = lane & 15;
    int m0 = wave * 16;
    prefetch_a(w3b, 128, m0, nl, half, 0);         // full 256B row in one call
    for (int j = 0; j < 32; ++j) {
        int idx = tid + j * 256;                   // 8192 elements
        int n = idx & 63, k = idx >> 6;
        Bt[n * 136 + k] = ub[(size_t)k * 16384 + n];
    }
    __syncthreads();
    v8f acc[4] = {};
    for (int s = 0; s < 4; ++s) {
        int kb = s * 32;
        v16bf a = load_a_global(w3b, 128, m0 + nl, kb, half);
#pragma unroll
        for (int g = 0; g < 4; ++g) {
            v16bf bfr = load_b_lds(Bt + (g * 16 + nl) * 136, kb, half);
            acc[g] = WMMA_BF16(a, bfr, acc[g]);
        }
    }
    size_t ob0 = (size_t)b * 2097152 + pix + nl;
#pragma unroll
    for (int r = 0; r < 8; ++r) {
        int m = m0 + r + 8 * half;
        float sa = s3a[m], ta = t3a[m], sb = s3b[m], tb2 = t3b[m];
        size_t mb = ob0 + (size_t)m * 16384;
#pragma unroll
        for (int g = 0; g < 4; ++g) {
            float v = fmaxf(sa * acc[g][r] + ta, 0.f);
            v = sb * v + tb2;
            v += ident[mb + g * 16];
            out[mb + g * 16] = fmaxf(v, 0.f);
        }
    }
}

extern "C" void kernel_launch(void* const* d_in, const int* in_sizes, int n_in,
                              void* d_out, int out_size, void* d_ws, size_t ws_size,
                              hipStream_t stream) {
    const float* x    = (const float*)d_in[0];
    const float* w1   = (const float*)d_in[1];
    const float* s1a  = (const float*)d_in[2];
    const float* t1a  = (const float*)d_in[3];
    const float* s1b  = (const float*)d_in[4];
    const float* t1b  = (const float*)d_in[5];
    const float* woff = (const float*)d_in[6];
    const float* boff = (const float*)d_in[7];
    const float* wdc  = (const float*)d_in[8];
    const float* bdc  = (const float*)d_in[9];
    const float* s2   = (const float*)d_in[10];
    const float* t2   = (const float*)d_in[11];
    const float* w3   = (const float*)d_in[12];
    const float* s3a  = (const float*)d_in[13];
    const float* t3a  = (const float*)d_in[14];
    const float* s3b  = (const float*)d_in[15];
    const float* t3b  = (const float*)d_in[16];
    const float* wds  = (const float*)d_in[17];
    const float* bds  = (const float*)d_in[18];
    const float* sd   = (const float*)d_in[19];
    const float* td   = (const float*)d_in[20];

    // workspace layout (all offsets 256B aligned)
    char* ws = (char*)d_ws;
    unsigned short* tbuf   = (unsigned short*)(ws);                  // 16,777,216 B
    unsigned short* ubuf   = (unsigned short*)(ws + 16777216);       // 16,777,216 B
    float*          identb = (float*)(ws + 33554432);                // 33,554,432 B
    float*          offbuf = (float*)(ws + 67108864);                //  7,077,888 B
    unsigned short* w1b    = (unsigned short*)(ws + 74186752);       //     65,536 B
    unsigned short* wdsb   = (unsigned short*)(ws + 74252288);       //     65,536 B
    unsigned short* woffb  = (unsigned short*)(ws + 74317824);       //     73,728 B
    unsigned short* wdcb   = (unsigned short*)(ws + 74391552);       //    294,912 B
    unsigned short* w3b    = (unsigned short*)(ws + 74686464);       //     32,768 B

    // weight prep (bf16 conversion + pad offset-conv weights 27 -> 32 rows)
    cvt_f32_bf16<<<(32768 + 255) / 256, 256, 0, stream>>>(w1, w1b, 32768);
    cvt_f32_bf16<<<(32768 + 255) / 256, 256, 0, stream>>>(wds, wdsb, 32768);
    cvt_f32_bf16<<<(31104 + 255) / 256, 256, 0, stream>>>(woff, woffb, 31104);
    fill_zero_u16<<<(5760 + 255) / 256, 256, 0, stream>>>(woffb + 31104, 36864 - 31104);
    cvt_f32_bf16<<<(147456 + 255) / 256, 256, 0, stream>>>(wdc, wdcb, 147456);
    cvt_f32_bf16<<<(16384 + 255) / 256, 256, 0, stream>>>(w3, w3b, 16384);

    // main pipeline: 65536 output pixels total
    k1<<<1024, 256, 0, stream>>>(x, w1b, wdsb, s1a, t1a, s1b, t1b, bds, sd, td,
                                 tbuf, identb);
    k2<<<1024, 256, 0, stream>>>(tbuf, woffb, boff, offbuf);
    k3<<<2048, 256, 0, stream>>>(tbuf, offbuf, wdcb, bdc, s2, t2, ubuf);
    k4<<<1024, 256, 0, stream>>>(ubuf, w3b, s3a, t3a, s3b, t3b, identb,
                                 (float*)d_out);
}
